// DifferentiableNeuralDictionary_88381837017632
// MI455X (gfx1250) — matrix operand order repaired and checked by reading, not compiled, not run on hardware
//
#include <hip/hip_runtime.h>

// ---------------------------------------------------------------------------
// DND lookup: d2 = ||q||^2 + ||e||^2 - 2 q.e ; dist = sqrt(max(d2,1e-12));
// top-10 smallest per query, denom = sum of ALL dists, out = sum((1-d)/denom * qv)
//
// Plan (MI455X / gfx1250):
//  1) prep kernel  : f32 -> f16 tables for queries+embeddings, norms from the
//                    rounded values, padded q-values/norms (zeros past N).
//  2) gemm kernel  : 128-query x 128-col tiles, v_wmma_f32_16x16x32_f16,
//                    double-buffered global_load_async_to_lds_b128 staging,
//                    on-the-fly denominator sum + threshold-guarded register
//                    top-10 per query (per N-slice).
//  3) merge kernel : fold 32 slice partials per query, emit output.
// Workspace required: ~33 MB.
// ---------------------------------------------------------------------------

#define Bq    2048
#define Nn    100000
#define Dd    128
#define Kk    10
#define NPAD  100096      // 782 chunks * 128
#define NSL   32          // N-slices (grid.y)
#define CHPS  25          // chunks (of 128 cols) per slice; last slice has 7
#define MBLK  128         // queries per workgroup (8 waves * 16)
#define CH    128         // embedding rows per staged chunk
#define EPITCH 136        // LDS emb row pitch in halves (272 B) - bank-conflict pad
#define DPITCH 20         // LDS dist row pitch in floats (80 B, 16B aligned)

typedef _Float16 v16h __attribute__((ext_vector_type(16)));
typedef _Float16 v8h  __attribute__((ext_vector_type(8)));
typedef float    v8f  __attribute__((ext_vector_type(8)));
typedef int      v4i  __attribute__((vector_size(4 * sizeof(int))));

typedef __attribute__((address_space(1))) v4i* gas_v4i_ptr;   // global
typedef __attribute__((address_space(3))) v4i* las_v4i_ptr;   // LDS

static __device__ inline v16h make16(v8h a, v8h b) {
  return __builtin_shufflevector(a, b, 0,1,2,3,4,5,6,7,8,9,10,11,12,13,14,15);
}

// Single-instruction v_sqrt_f32 (<=2 ulp) -- the f16 WMMA path already has
// ~1e-3 relative error, so the IEEE-exact libm expansion is wasted VALU.
static __device__ inline float fast_sqrtf(float x) {
#if __has_builtin(__builtin_amdgcn_sqrtf)
  return __builtin_amdgcn_sqrtf(x);
#else
  return sqrtf(x);
#endif
}

// ---- CDNA5 async global->LDS copy (ASYNCcnt-tracked) -----------------------
// Pointers are rebuilt via integers: global flat address == AS1 address, and
// the low 32 bits of a generic LDS address are the AS3 offset (aperture map).
static __device__ inline void cp_async16(const void* g, void* l) {
#if __has_builtin(__builtin_amdgcn_global_load_async_to_lds_b128)
  __builtin_amdgcn_global_load_async_to_lds_b128(
      (gas_v4i_ptr)(unsigned long long)g,
      (las_v4i_ptr)(unsigned)(unsigned long long)l, 0, 0);
#else
  unsigned loff = (unsigned)(unsigned long long)l;  // LDS aperture: addr[31:0]
  asm volatile("global_load_async_to_lds_b128 %0, %1, off"
               :: "v"(loff), "v"(g) : "memory");
#endif
}

#if __has_builtin(__builtin_amdgcn_s_wait_asynccnt)
#define WAIT_ASYNC(imm) __builtin_amdgcn_s_wait_asynccnt(imm)
#else
#define WAIT_ASYNC(imm) asm volatile("s_wait_asynccnt " #imm ::: "memory")
#endif

// ---- branchless register top-K (replace current max) -----------------------
static __device__ inline void topk_insert(float (&td)[Kk], float (&tq)[Kk],
                                          float d, float q, float& thr) {
  float dmax = td[0];
#pragma unroll
  for (int i = 1; i < Kk; ++i) dmax = fmaxf(dmax, td[i]);
  bool done = false;
#pragma unroll
  for (int i = 0; i < Kk; ++i) {
    if (!done && td[i] == dmax) { td[i] = d; tq[i] = q; done = true; }
  }
  float m = td[0];
#pragma unroll
  for (int i = 1; i < Kk; ++i) m = fmaxf(m, td[i]);
  thr = m;
}

// ---------------------------------------------------------------------------
// Kernel 1: convert rows to f16, norms from rounded values, padded copies.
// ---------------------------------------------------------------------------
static __device__ inline float cvt_row(const float* __restrict__ src,
                                       _Float16* __restrict__ dst) {
  float s = 0.f;
#pragma unroll 1
  for (int i = 0; i < 16; ++i) {
    float4 a = ((const float4*)src)[2 * i];
    float4 b = ((const float4*)src)[2 * i + 1];
    v8h h;
    h[0] = (_Float16)a.x; h[1] = (_Float16)a.y; h[2] = (_Float16)a.z; h[3] = (_Float16)a.w;
    h[4] = (_Float16)b.x; h[5] = (_Float16)b.y; h[6] = (_Float16)b.z; h[7] = (_Float16)b.w;
#pragma unroll
    for (int e = 0; e < 8; ++e) { float v = (float)h[e]; s += v * v; }
    ((v8h*)dst)[i] = h;
  }
  return s;
}

__global__ __launch_bounds__(256) void dnd_prep(
    const float* __restrict__ st, const float* __restrict__ em,
    const float* __restrict__ qv,
    _Float16* __restrict__ ef16, _Float16* __restrict__ qf16,
    float* __restrict__ enp, float* __restrict__ qnp, float* __restrict__ qvp) {
  const long r = (long)blockIdx.x * 256 + threadIdx.x;
  if (r < Nn) {
    enp[r] = cvt_row(em + r * Dd, ef16 + r * Dd);
    qvp[r] = qv[r];
  } else if (r < NPAD) {                 // zero pad rows (masked in epilogue)
    v8h z = {};
#pragma unroll 1
    for (int i = 0; i < 16; ++i) ((v8h*)(ef16 + r * Dd))[i] = z;
    enp[r] = 0.f; qvp[r] = 0.f;
  } else if (r < NPAD + Bq) {
    const long q = r - NPAD;
    qnp[q] = cvt_row(st + q * Dd, qf16 + q * Dd);
  }
}

// ---------------------------------------------------------------------------
// Kernel 2: WMMA GEMM + fused sum / top-K per N-slice.
// ---------------------------------------------------------------------------
static __device__ inline void issue_chunk(int tid, int gRow0,
    const _Float16* __restrict__ ef16, const float* __restrict__ enp,
    const float* __restrict__ qvp,
    _Float16* sEmb, float* sEn, float* sQv) {
#pragma unroll
  for (int i = 0; i < 8; ++i) {              // 128 rows * 256B, 8 x 16B / thread
    int g = i * 256 + tid;                   // 0..2047 16B-chunks
    int row = g >> 4, sub = g & 15;
    const void* src = ef16 + (size_t)(gRow0 + row) * Dd + sub * 8;
    void* dst = (char*)sEmb + row * (EPITCH * 2) + sub * 16;
    cp_async16(src, dst);
  }
  if (tid < 32)                              // ||e||^2 chunk (512 B)
    cp_async16(enp + gRow0 + tid * 4, (char*)sEn + tid * 16);
  else if (tid < 64)                         // q-values chunk (512 B)
    cp_async16(qvp + gRow0 + (tid - 32) * 4, (char*)sQv + (tid - 32) * 16);
}

__global__ __launch_bounds__(256) void dnd_gemm(
    const _Float16* __restrict__ ef16, const _Float16* __restrict__ qf16,
    const float* __restrict__ enp, const float* __restrict__ qnp,
    const float* __restrict__ qvp,
    float* __restrict__ pSum, float* __restrict__ pD, float* __restrict__ pQ) {
  __shared__ __align__(16) _Float16 sEmb[2][CH * EPITCH];   // 2 x 34 KB
  __shared__ __align__(16) float    sEn[2][CH];
  __shared__ __align__(16) float    sQv[2][CH];
  __shared__ __align__(16) float    sDist[8][16 * DPITCH];  // per-wave transpose

  const int tid  = threadIdx.x;
  const int lane = tid & 31;
  const int l16  = lane & 15;
  const bool loh = lane < 16;
  const int wv   = tid >> 5;
  const int mb   = blockIdx.x, sl = blockIdx.y;

  const int colBeg = sl * (CHPS * CH);
  int colEnd = colBeg + CHPS * CH; if (colEnd > Nn) colEnd = Nn;
  const int nch = (colEnd - colBeg + CH - 1) / CH;

  // --- resident A tile: 16 query rows, 4 K-chunks (documented f16 A layout) --
  const int mrow = mb * MBLK + wv * 16;
  v16h A[4];
  {
    const _Float16* qr = qf16 + (size_t)(mrow + l16) * Dd;
    const int koff = loh ? 0 : 8;
#pragma unroll
    for (int k = 0; k < 4; ++k) {
      v8h a0 = *(const v8h*)(qr + 32 * k + koff);        // K = 32k+koff .. +7
      v8h a1 = *(const v8h*)(qr + 32 * k + 16 + koff);   // K = 32k+16+koff ..
      A[k] = make16(a0, a1);
    }
  }
  float qnr[8];
#pragma unroll
  for (int r = 0; r < 8; ++r) qnr[r] = qnp[mrow + r + (loh ? 0 : 8)];

  float td[Kk], tq[Kk];
#pragma unroll
  for (int i = 0; i < Kk; ++i) { td[i] = 3.0e38f; tq[i] = 0.f; }
  float thr = 3.0e38f, rsum = 0.f;

  issue_chunk(tid, colBeg, ef16, enp, qvp, sEmb[0], sEn[0], sQv[0]);

  for (int c = 0; c < nch; ++c) {
    const int p = c & 1;
    __syncthreads();                                   // buf[1-p] free to refill
    const bool pref = (c + 1) < nch;
    if (pref)
      issue_chunk(tid, colBeg + (c + 1) * CH, ef16, enp, qvp,
                  sEmb[1 - p], sEn[1 - p], sQv[1 - p]);
    if (pref) { if (tid < 64) WAIT_ASYNC(9); else WAIT_ASYNC(8); }
    else      { WAIT_ASYNC(0); }
    __syncthreads();                                   // chunk c fully in LDS

    const _Float16* eb = sEmb[p];
    const float* en  = sEn[p];
    const float* qvs = sQv[p];
    const int c0 = colBeg + c * CH;

#pragma unroll 1
    for (int t = 0; t < 8; ++t) {                      // 8 col tiles of 16
      const _Float16* brow = eb + (t * 16 + l16) * EPITCH + (loh ? 0 : 8);
      v8f acc = {};
#pragma unroll
      for (int k = 0; k < 4; ++k) {
        v8h b0 = *(const v8h*)(brow + 32 * k);
        v8h b1 = *(const v8h*)(brow + 32 * k + 16);
        acc = __builtin_amdgcn_wmma_f32_16x16x32_f16(
            false, A[k], false, make16(b0, b1), (short)0, acc, false, false);
      }
      // distances + LDS transpose (same-wave DS ops are in order)
      const float enl = en[t * 16 + l16];
      float* db = &sDist[wv][0];
#pragma unroll
      for (int r = 0; r < 8; ++r) {
        float d2 = qnr[r] + enl - 2.0f * acc[r];
        float dd = fast_sqrtf(fmaxf(d2, 1e-12f));
        db[(r + (loh ? 0 : 8)) * DPITCH + l16] = dd;
      }
      if (loh) {                                       // lane owns query row
        const float4* rp = (const float4*)&sDist[wv][lane * DPITCH];
        const int nb = c0 + t * 16;
#pragma unroll 1
        for (int jj = 0; jj < 4; ++jj) {
          float4 f4 = rp[jj];
          float fe[4] = {f4.x, f4.y, f4.z, f4.w};
#pragma unroll
          for (int e = 0; e < 4; ++e) {
            const int n = nb + jj * 4 + e;
            const bool valid = n < colEnd;
            float dd = fe[e];
            rsum += valid ? dd : 0.f;
            if (valid && dd < thr)
              topk_insert(td, tq, dd, qvs[t * 16 + jj * 4 + e], thr);
          }
        }
      }
    }
  }

  if (loh) {                                           // per-(slice,query) partials
    const int b = mrow + lane;
    const size_t base = (size_t)sl * Bq + b;
    pSum[base] = rsum;
#pragma unroll
    for (int i = 0; i < Kk; ++i) { pD[base * Kk + i] = td[i]; pQ[base * Kk + i] = tq[i]; }
  }
}

// ---------------------------------------------------------------------------
// Kernel 3: merge slice partials, emit output.
// ---------------------------------------------------------------------------
__global__ __launch_bounds__(256) void dnd_merge(
    const float* __restrict__ ps, const float* __restrict__ pd,
    const float* __restrict__ pq, float* __restrict__ out) {
  const int b = blockIdx.x * 256 + threadIdx.x;
  if (b >= Bq) return;
  float denom = 0.f;
  float td[Kk], tq[Kk];
#pragma unroll
  for (int i = 0; i < Kk; ++i) { td[i] = 3.0e38f; tq[i] = 0.f; }
  float thr = 3.0e38f;
  for (int s = 0; s < NSL; ++s) {
    denom += ps[(size_t)s * Bq + b];
    const size_t base = ((size_t)s * Bq + b) * Kk;
#pragma unroll 1
    for (int i = 0; i < Kk; ++i) {
      float d = pd[base + i];
      if (d < thr) topk_insert(td, tq, d, pq[base + i], thr);
    }
  }
  float acc = 0.f;
#pragma unroll
  for (int i = 0; i < Kk; ++i)
    if (td[i] < 3.0e38f) acc += (1.0f - td[i]) * tq[i];
  out[b] = acc / denom;
}

// ---------------------------------------------------------------------------
extern "C" void kernel_launch(void* const* d_in, const int* in_sizes, int n_in,
                              void* d_out, int out_size, void* d_ws, size_t ws_size,
                              hipStream_t stream) {
  (void)in_sizes; (void)n_in; (void)out_size; (void)ws_size;
  const float* st = (const float*)d_in[0];   // state      [2048,128]
  const float* em = (const float*)d_in[1];   // embeddings [100000,128]
  const float* qv = (const float*)d_in[2];   // q_values   [100000]

  char* ws = (char*)d_ws;
  size_t off = 0;
  auto alloc = [&](size_t bytes) -> char* {
    char* p = ws + off; off += (bytes + 255) & ~(size_t)255; return p;
  };
  _Float16* ef16 = (_Float16*)alloc((size_t)NPAD * Dd * 2);   // 25.6 MB
  _Float16* qf16 = (_Float16*)alloc((size_t)Bq * Dd * 2);     // 0.5 MB
  float* enp = (float*)alloc((size_t)NPAD * 4);
  float* qvp = (float*)alloc((size_t)NPAD * 4);
  float* qnp = (float*)alloc((size_t)Bq * 4);
  float* ps  = (float*)alloc((size_t)NSL * Bq * 4);
  float* pd  = (float*)alloc((size_t)NSL * Bq * Kk * 4);
  float* pq  = (float*)alloc((size_t)NSL * Bq * Kk * 4);      // total ~33 MB

  dnd_prep<<<(NPAD + Bq + 255) / 256, 256, 0, stream>>>(
      st, em, qv, ef16, qf16, enp, qnp, qvp);
  dnd_gemm<<<dim3(Bq / MBLK, NSL), 256, 0, stream>>>(
      ef16, qf16, enp, qnp, qvp, ps, pd, pq);
  dnd_merge<<<(Bq + 255) / 256, 256, 0, stream>>>(ps, pd, pq, (float*)d_out);
}
